// SSIMLoss_91336774517073
// MI455X (gfx1250) — matrix operand order repaired and checked by reading, not compiled, not run on hardware
//
#include <hip/hip_runtime.h>

typedef __attribute__((ext_vector_type(16))) _Float16 v16h;
typedef __attribute__((ext_vector_type(2)))  _Float16 v2h;
typedef __attribute__((ext_vector_type(8)))  float    v8f;

#define IMG_H 512
#define IMG_W 512
#define IN_ROWS 32
#define IN_COLS 144
#define IN_STRIDE 148   // padded f16 stride -> conflict-free row access
#define H_STRIDE 34     // padded f16 stride for H^T columns

// Normalized 11-tap Gaussian, sigma=1.5 (matches reference window)
constexpr float GWC[11] = {
    0.0010285f, 0.0075988f, 0.0360007f, 0.1093606f, 0.2130054f,
    0.2660117f, 0.2130054f, 0.1093606f, 0.0360007f, 0.0075988f,
    0.0010285f};

constexpr float gwf(int k) { return (k >= 0 && k < 11) ? GWC[k] : 0.0f; }

// Per-lane pre-swizzled band operands, computed at compile time.
//   b[lane][.] : B operand (32x16) for the horizontal pass
//                lane: n = lane&15, kh = lane>>4; dword v holds rows
//                K = (v<4 ? 2v : 2v+8) + 8*kh, K+1 at column n.
//   a[lane][.] : A operand (16x32) for the vertical pass
//                lane: m = lane&15, kh = lane>>4; dword i holds
//                K = 2i + 16*kh, K+1 for row m.
// band(j, n) = g[j - n - 1] (zero outside taps); the -1 comes from the
// tile halo being shifted by -6 so packed f16 pairs stay dword aligned.
struct alignas(32) BandTbl {
    _Float16 b[32][16];
    _Float16 a[32][16];
};

constexpr BandTbl make_band() {
    BandTbl t{};
    for (int lane = 0; lane < 32; ++lane) {
        const int n  = lane & 15;
        const int kh = lane >> 4;
        for (int v = 0; v < 8; ++v) {
            const int K = (v < 4 ? 2 * v : 2 * v + 8) + 8 * kh;
            t.b[lane][2 * v]     = (_Float16)gwf(K - n - 1);
            t.b[lane][2 * v + 1] = (_Float16)gwf(K - n);
        }
        for (int i = 0; i < 8; ++i) {
            const int K = 2 * i + 16 * kh;
            t.a[lane][2 * i]     = (_Float16)gwf(K - n - 1);
            t.a[lane][2 * i + 1] = (_Float16)gwf(K - n);
        }
    }
    return t;
}

__device__ constexpr BandTbl BAND = make_band();

union AU {
    v16h h;
    v2h  p[8];
};

static __device__ __forceinline__ v8f wmma16(v16h a, v16h b, v8f c) {
    return __builtin_amdgcn_wmma_f32_16x16x32_f16(
        false, a, false, b, (short)0, c, false, false);
}

// Spill a 16x16 f32 D half-tile to LDS as transposed f16 (H^T layout).
// D layout: lane l, vgpr v  ->  row = v + 8*(l/16), col = l%16.
static __device__ __forceinline__ void spillH(v8f d, _Float16* base,
                                              int colN, int rbase) {
    _Float16* q = base + colN * H_STRIDE + rbase;
#pragma unroll
    for (int t = 0; t < 4; ++t) {
        v2h pk;
        pk[0] = (_Float16)d[2 * t];
        pk[1] = (_Float16)d[2 * t + 1];
        *(v2h*)(q + 2 * t) = pk;
    }
}

// Vertical pass: D = bandA(16x32) x H(32x16).  B layout: lane l, vgpr v
// holds K = 2v + 16*(l/16), K+1 at column N = l%16.
static __device__ __forceinline__ v8f vpass(const _Float16* base, int colN,
                                            int kh, v16h bandA) {
    AU b;
#pragma unroll
    for (int v = 0; v < 8; ++v)
        b.p[v] = *(const v2h*)(base + colN * H_STRIDE + 2 * v + 16 * kh);
    v8f z = {};
    return wmma16(bandA, b.h, z);
}

__global__ __launch_bounds__(256) void ssim_main(const float* __restrict__ img1,
                                                 const float* __restrict__ img2,
                                                 float* __restrict__ accum) {
    __shared__ _Float16 sIn1[IN_ROWS][IN_STRIDE];
    __shared__ _Float16 sIn2[IN_ROWS][IN_STRIDE];
    __shared__ _Float16 sH[8][5][16][H_STRIDE];
    __shared__ float    sRed[8];

    const int tid  = threadIdx.x;
    const int w    = tid >> 5;   // wave id (wave32)
    const int lane = tid & 31;
    const int m16  = lane & 15;
    const int kh   = lane >> 4;

    const int plane = blockIdx.z;
    const int r0    = blockIdx.y * 16;
    const int c0    = blockIdx.x * 128;
    const float* p1 = img1 + (size_t)plane * (IMG_H * IMG_W);
    const float* p2 = img2 + (size_t)plane * (IMG_H * IMG_W);

    // ---- band operands: two 32B loads each from compile-time table ----
    const v16h bandB = *(const v16h*)(&BAND.b[lane][0]);
    const v16h bandA = *(const v16h*)(&BAND.a[lane][0]);

    // ---- cooperative zero-padded load: global f32 -> LDS f16 ----
    // tid -> (row = tid>>4, col = tid&15); 2 row-halves x 9 col chunks,
    // shift/add indexing only, coalesced 16-wide.
    {
        const int ccol = tid & 15;
        const int irow = tid >> 4;
#pragma unroll
        for (int half = 0; half < 2; ++half) {
            const int i   = irow + 16 * half;
            const int row = r0 - 6 + i;
            const bool rok = (row >= 0) && (row < IMG_H);
            const float* g1 = p1 + row * IMG_W;
            const float* g2 = p2 + row * IMG_W;
#pragma unroll
            for (int ch = 0; ch < 9; ++ch) {
                const int cc  = ch * 16 + ccol;
                const int col = c0 - 6 + cc;
                const bool ok = rok && (col >= 0) && (col < IMG_W);
                float a = ok ? g1[col] : 0.0f;
                float b = ok ? g2[col] : 0.0f;
                sIn1[i][cc] = (_Float16)a;
                sIn2[i][cc] = (_Float16)b;
            }
        }
    }
    __syncthreads();

    // ---- horizontal pass: 2 row-halves x 5 quantities, spill H^T ----
    const int cbase = w * 16;
#pragma unroll
    for (int p = 0; p < 2; ++p) {
        const int lrow = p * 16 + m16;  // A row = extended image row
        const _Float16* r1 = &sIn1[lrow][cbase];
        const _Float16* r2 = &sIn2[lrow][cbase];
        AU ax, ay, axx, ayy, axy;
#pragma unroll
        for (int i = 0; i < 8; ++i) {
            int K = (i < 4 ? 2 * i : 2 * i + 8) + 8 * kh;
            v2h x = *(const v2h*)(r1 + K);
            v2h y = *(const v2h*)(r2 + K);
            ax.p[i]  = x;
            ay.p[i]  = y;
            axx.p[i] = x * x;   // v_pk_mul_f16
            ayy.p[i] = y * y;
            axy.p[i] = x * y;
        }
        v8f z = {};
        v8f h0 = wmma16(ax.h,  bandB, z);
        v8f h1 = wmma16(ay.h,  bandB, z);
        v8f h2 = wmma16(axx.h, bandB, z);
        v8f h3 = wmma16(ayy.h, bandB, z);
        v8f h4 = wmma16(axy.h, bandB, z);
        const int rbase = 8 * kh + 16 * p;
        spillH(h0, &sH[w][0][0][0], m16, rbase);
        spillH(h1, &sH[w][1][0][0], m16, rbase);
        spillH(h2, &sH[w][2][0][0], m16, rbase);
        spillH(h3, &sH[w][3][0][0], m16, rbase);
        spillH(h4, &sH[w][4][0][0], m16, rbase);
    }
    // per-wave LDS ops are in order (DScnt); sH is private to each wave

    // ---- vertical pass: 5 WMMAs ----
    v8f dmu1 = vpass(&sH[w][0][0][0], m16, kh, bandA);
    v8f dmu2 = vpass(&sH[w][1][0][0], m16, kh, bandA);
    v8f dxx  = vpass(&sH[w][2][0][0], m16, kh, bandA);
    v8f dyy  = vpass(&sH[w][3][0][0], m16, kh, bandA);
    v8f dxy  = vpass(&sH[w][4][0][0], m16, kh, bandA);

    // ---- elementwise SSIM + reduction ----
    float acc = 0.0f;
#pragma unroll
    for (int v = 0; v < 8; ++v) {
        float mu1 = dmu1[v], mu2 = dmu2[v];
        float m11 = mu1 * mu1, m22 = mu2 * mu2, m12 = mu1 * mu2;
        float s1  = fmaxf(dxx[v] - m11, 0.0f);
        float s2  = fmaxf(dyy[v] - m22, 0.0f);
        float s12 = dxy[v] - m12;
        const float C1 = 1e-4f, C2 = 9e-4f;
        float num = (2.0f * m12 + C1) * (2.0f * s12 + C2);
        float den = (m11 + m22 + C1) * (s1 + s2 + C2);
        acc += num / den;
    }
#pragma unroll
    for (int off = 16; off; off >>= 1) acc += __shfl_xor(acc, off, 32);
    if (lane == 0) sRed[w] = acc;
    __syncthreads();
    if (tid == 0) {
        float s = 0.0f;
#pragma unroll
        for (int i = 0; i < 8; ++i) s += sRed[i];
        atomicAdd(accum, s);
    }
}

__global__ void ssim_zero(float* accum) {
    if (threadIdx.x == 0) accum[0] = 0.0f;
}

__global__ void ssim_final(const float* __restrict__ accum,
                           float* __restrict__ out) {
    if (threadIdx.x == 0)
        out[0] = 1.0f - accum[0] * (1.0f / 25165824.0f);  // 32*3*512*512
}

extern "C" void kernel_launch(void* const* d_in, const int* in_sizes, int n_in,
                              void* d_out, int out_size, void* d_ws,
                              size_t ws_size, hipStream_t stream) {
    (void)in_sizes; (void)n_in; (void)out_size; (void)ws_size;
    const float* img1 = (const float*)d_in[0];
    const float* img2 = (const float*)d_in[1];
    float* out   = (float*)d_out;
    float* accum = (float*)d_ws;

    ssim_zero<<<1, 32, 0, stream>>>(accum);
    dim3 grid(4, 32, 96);  // 128-col blocks x 16-row blocks x (N*C) planes
    ssim_main<<<grid, 256, 0, stream>>>(img1, img2, accum);
    ssim_final<<<1, 32, 0, stream>>>(accum, out);
}
